// NeuralCF_24197845745667
// MI455X (gfx1250) — matrix operand order repaired and checked by reading, not compiled, not run on hardware
//
#include <hip/hip_runtime.h>
#include <hip/hip_bf16.h>
#include <stdint.h>

#define NNODES 100000
#define NPAD   100352      // 6272 tiles = 196 blocks x 32 tiles (padded; pad rows never read)
#define EMBD   128
#define NEDGE  640000
#define BPAIRS 16384

typedef __attribute__((ext_vector_type(16))) __bf16        v16bf;
typedef __attribute__((ext_vector_type(8)))  float         v8f;
typedef __attribute__((ext_vector_type(4)))  unsigned int  u32x4;
typedef __attribute__((ext_vector_type(2)))  unsigned int  u32x2;
typedef __attribute__((ext_vector_type(4)))  float         f32x4;

union Frag16 { v16bf v; u32x4 q[2]; };

__device__ __forceinline__ unsigned short f2bf(float f) {
  unsigned int u = __float_as_uint(f);
  u += 0x7FFFu + ((u >> 16) & 1u);          // round-to-nearest-even
  return (unsigned short)(u >> 16);
}

// ---------------------------------------------------------------- convert f32 -> bf16
__global__ void __launch_bounds__(256)
f32_to_bf16(const float* __restrict__ in, unsigned short* __restrict__ out, int n) {
  int i = blockIdx.x * 256 + threadIdx.x;
  if (i < n) out[i] = f2bf(in[i]);
}

// ---------------------------------------------------------------- WRGCN dense transforms
// Block = 8 waves; all 3 weight matrices (Wr0|Wr1|Wroot, 96KB bf16) staged in LDS once.
// Each wave owns FOUR 16-node M tiles: every LDS B fragment feeds 4 independent WMMAs.
// Node dim padded to NPAD so there are NO guards -> EXEC is all-1s for every WMMA.
__global__ void __launch_bounds__(256)
wrgcn_gemm(const unsigned short* __restrict__ xb,    // [NPAD,128] bf16
           const unsigned short* __restrict__ Wall,  // [3,128,128] bf16: Wr0,Wr1,Wroot
           const float* __restrict__ bias,           // [128]
           float* __restrict__ h,                    // [2,NPAD,128] f32
           float* __restrict__ agg)                  // [NPAD,128] f32
{
  __shared__ __align__(16) unsigned short sW[3 * EMBD * EMBD];   // 96 KB
  {
    const u32x4* s4 = (const u32x4*)Wall;
    u32x4*       d4 = (u32x4*)sW;
#pragma unroll 4
    for (int i = threadIdx.x; i < (3 * EMBD * EMBD) / 8; i += 256) d4[i] = s4[i];
  }
  __syncthreads();

  const int lane = threadIdx.x & 31;
  const int wid  = threadIdx.x >> 5;
  const int t0   = blockIdx.x * 32 + wid * 4;        // four M tiles per wave
  const int m    = lane & 15;
  const int half = lane >> 4;
  const int krow = m + (half << 4);                  // 0..31 inside a K=32 tile (B layout)

  // A fragments: four 16x128 bf16 row blocks resident in VGPRs (128 VGPRs)
  Frag16 A[4][4];
#pragma unroll
  for (int t = 0; t < 4; ++t) {
    const unsigned short* ar = xb + ((size_t)(t0 + t) * 16 + m) * EMBD;
#pragma unroll
    for (int kt = 0; kt < 4; ++kt) {
      const unsigned short* p = ar + kt * 32 + half * 8;
      A[t][kt].q[0] = *(const u32x4*)(p);            // K = base + half*8 + 0..7
      A[t][kt].q[1] = *(const u32x4*)(p + 16);       // K = base + 16 + half*8 + 0..7
    }
  }

#pragma unroll
  for (int w = 0; w < 3; ++w) {
    float* outBase = (w == 2) ? agg : (h + (size_t)w * NPAD * EMBD);
#pragma unroll
    for (int nt = 0; nt < 8; ++nt) {
      const float binit = (w == 2) ? bias[nt * 16 + m] : 0.0f;
      v8f acc[4];
#pragma unroll
      for (int t = 0; t < 4; ++t)
#pragma unroll
        for (int i = 0; i < 8; ++i) acc[t][i] = binit;
#pragma unroll
      for (int kt = 0; kt < 4; ++kt) {
        Frag16 Bf;
        const unsigned short* bp = sW + w * EMBD * EMBD
                                      + (size_t)(kt * 32) * EMBD + nt * 16 + m;
#pragma unroll
        for (int j = 0; j < 8; ++j) {
          const int k0 = half * 8 + ((2 * j) & 7) + (((2 * j) >> 3) << 4);
          Bf.q[j >> 2][j & 3] = (unsigned int)bp[k0 * EMBD] | ((unsigned int)bp[(k0 + 1) * EMBD] << 16);
        }
#pragma unroll
        for (int t = 0; t < 4; ++t)
          acc[t] = __builtin_amdgcn_wmma_f32_16x16x32_bf16(false, A[t][kt].v, false, Bf.v,
                                                           (short)0, acc[t], false, false);
      }
#pragma unroll
      for (int t = 0; t < 4; ++t) {
        const int rbase = (t0 + t) * 16 + half * 8;
#pragma unroll
        for (int i = 0; i < 8; ++i)
          outBase[(size_t)(rbase + i) * EMBD + nt * 16 + m] = acc[t][i];
      }
    }
  }
}

// ---------------------------------------------------------------- edge scatter (segment_sum)
// one wave per edge, 4 floats per lane; working set (h 103MB + agg 51MB) is L2-resident
__global__ void __launch_bounds__(256)
edge_scatter(const int* __restrict__ src, const int* __restrict__ dst,
             const int* __restrict__ et,  const float* __restrict__ ew,
             const float* __restrict__ h, float* __restrict__ agg, int nE)
{
  const int e = blockIdx.x * 8 + (threadIdx.x >> 5);
  if (e >= nE) return;
  const int lane = threadIdx.x & 31;
  const int s = src[e], d = dst[e], r = et[e];
  const float w = ew[e];
  f32x4 msg = *((const f32x4*)(h + ((size_t)r * NPAD + s) * EMBD) + lane);
  float* ap = agg + (size_t)d * EMBD + lane * 4;
  atomicAdd(ap + 0, w * msg.x);
  atomicAdd(ap + 1, w * msg.y);
  atomicAdd(ap + 2, w * msg.z);
  atomicAdd(ap + 3, w * msg.w);
}

// ---------------------------------------------------------------- relu + layernorm -> bf16
__global__ void __launch_bounds__(256)
relu_layernorm_bf16(const float* __restrict__ agg,
                    const float* __restrict__ g, const float* __restrict__ b,
                    unsigned short* __restrict__ xb, int nNodes)
{
  const int n = blockIdx.x * 8 + (threadIdx.x >> 5);
  if (n >= nNodes) return;
  const int lane = threadIdx.x & 31;
  f32x4 y = *((const f32x4*)(agg + (size_t)n * EMBD) + lane);
  y.x = fmaxf(y.x, 0.f); y.y = fmaxf(y.y, 0.f);
  y.z = fmaxf(y.z, 0.f); y.w = fmaxf(y.w, 0.f);
  float s = y.x + y.y + y.z + y.w;
#pragma unroll
  for (int o = 16; o >= 1; o >>= 1) s += __shfl_xor(s, o, 32);
  const float mu = s * (1.0f / EMBD);
  f32x4 dx; dx.x = y.x - mu; dx.y = y.y - mu; dx.z = y.z - mu; dx.w = y.w - mu;
  float v = dx.x * dx.x + dx.y * dx.y + dx.z * dx.z + dx.w * dx.w;
#pragma unroll
  for (int o = 16; o >= 1; o >>= 1) v += __shfl_xor(v, o, 32);
  const float inv = rsqrtf(v * (1.0f / EMBD) + 1e-5f);
  f32x4 gg = *((const f32x4*)g + lane);
  f32x4 bb = *((const f32x4*)b + lane);
  unsigned int o0 = f2bf(dx.x * inv * gg.x + bb.x);
  unsigned int o1 = f2bf(dx.y * inv * gg.y + bb.y);
  unsigned int o2 = f2bf(dx.z * inv * gg.z + bb.z);
  unsigned int o3 = f2bf(dx.w * inv * gg.w + bb.w);
  u32x2 pk; pk.x = o0 | (o1 << 16); pk.y = o2 | (o3 << 16);
  *((u32x2*)(xb + (size_t)n * EMBD) + lane) = pk;
}

// ---------------------------------------------------------------- head: gather + MLP (WMMA) + GMF + sigmoid
__global__ void __launch_bounds__(32)
neuralcf_head(const float* __restrict__ x3,
              const int* __restrict__ user, const int* __restrict__ item,
              const unsigned short* __restrict__ W1b, const float* __restrict__ b1,
              const unsigned short* __restrict__ W2b, const float* __restrict__ b2,
              const unsigned short* __restrict__ W3b, const float* __restrict__ b3,
              const float* __restrict__ Wout, const float* __restrict__ bout,
              float* __restrict__ out)
{
  __shared__ __align__(16) unsigned short sA[16 * 256];
  __shared__ __align__(16) unsigned short sH1[16 * 128];
  __shared__ __align__(16) unsigned short sH2[16 * 64];
  __shared__ __align__(16) float          sH3[16 * 32];

  const int lane = threadIdx.x & 31;
  const int base = blockIdx.x * 16;
  const int m = lane & 15, half = lane >> 4;
  const int krow = m + (half << 4);

  // gather concat(u,v) as bf16 into LDS (row-major [16][256])
  for (int p = 0; p < 16; ++p) {
    const float* up = x3 + (size_t)user[base + p] * EMBD;
    const float* vp = x3 + (size_t)item[base + p] * EMBD;
    const int c0 = lane * 8;
#pragma unroll
    for (int k = 0; k < 8; ++k) {
      int c = c0 + k;
      sA[p * 256 + c] = f2bf((c < EMBD) ? up[c] : vp[c - EMBD]);
    }
  }
  __syncthreads();

  // MLP1: [16,256]x[256,128]
  {
    Frag16 A[8];
    const unsigned short* arow = sA + m * 256;
#pragma unroll
    for (int kt = 0; kt < 8; ++kt) {
      const unsigned short* p = arow + kt * 32 + half * 8;
      A[kt].q[0] = *(const u32x4*)(p);
      A[kt].q[1] = *(const u32x4*)(p + 16);
    }
#pragma unroll
    for (int nt = 0; nt < 8; ++nt) {
      v8f acc; const float bi = b1[nt * 16 + m];
#pragma unroll
      for (int i = 0; i < 8; ++i) acc[i] = bi;
#pragma unroll
      for (int kt = 0; kt < 8; ++kt) {
        Frag16 Bf;
        const unsigned short* bp = W1b + (size_t)(kt * 32) * 128 + nt * 16 + m;
#pragma unroll
        for (int j = 0; j < 8; ++j) {
          const int k0 = half * 8 + ((2 * j) & 7) + (((2 * j) >> 3) << 4);
          Bf.q[j >> 2][j & 3] = (unsigned int)bp[k0 * 128] | ((unsigned int)bp[(k0 + 1) * 128] << 16);
        }
        acc = __builtin_amdgcn_wmma_f32_16x16x32_bf16(false, A[kt].v, false, Bf.v,
                                                      (short)0, acc, false, false);
      }
#pragma unroll
      for (int i = 0; i < 8; ++i)
        sH1[(half * 8 + i) * 128 + nt * 16 + m] = f2bf(fmaxf(acc[i], 0.f));
    }
  }
  __syncthreads();

  // MLP2: [16,128]x[128,64]
  {
    Frag16 A[4];
    const unsigned short* arow = sH1 + m * 128;
#pragma unroll
    for (int kt = 0; kt < 4; ++kt) {
      const unsigned short* p = arow + kt * 32 + half * 8;
      A[kt].q[0] = *(const u32x4*)(p);
      A[kt].q[1] = *(const u32x4*)(p + 16);
    }
#pragma unroll
    for (int nt = 0; nt < 4; ++nt) {
      v8f acc; const float bi = b2[nt * 16 + m];
#pragma unroll
      for (int i = 0; i < 8; ++i) acc[i] = bi;
#pragma unroll
      for (int kt = 0; kt < 4; ++kt) {
        Frag16 Bf;
        const unsigned short* bp = W2b + (size_t)(kt * 32) * 64 + nt * 16 + m;
#pragma unroll
        for (int j = 0; j < 8; ++j) {
          const int k0 = half * 8 + ((2 * j) & 7) + (((2 * j) >> 3) << 4);
          Bf.q[j >> 2][j & 3] = (unsigned int)bp[k0 * 64] | ((unsigned int)bp[(k0 + 1) * 64] << 16);
        }
        acc = __builtin_amdgcn_wmma_f32_16x16x32_bf16(false, A[kt].v, false, Bf.v,
                                                      (short)0, acc, false, false);
      }
#pragma unroll
      for (int i = 0; i < 8; ++i)
        sH2[(half * 8 + i) * 64 + nt * 16 + m] = f2bf(fmaxf(acc[i], 0.f));
    }
  }
  __syncthreads();

  // MLP3: [16,64]x[64,32]
  {
    Frag16 A[2];
    const unsigned short* arow = sH2 + m * 64;
#pragma unroll
    for (int kt = 0; kt < 2; ++kt) {
      const unsigned short* p = arow + kt * 32 + half * 8;
      A[kt].q[0] = *(const u32x4*)(p);
      A[kt].q[1] = *(const u32x4*)(p + 16);
    }
#pragma unroll
    for (int nt = 0; nt < 2; ++nt) {
      v8f acc; const float bi = b3[nt * 16 + m];
#pragma unroll
      for (int i = 0; i < 8; ++i) acc[i] = bi;
#pragma unroll
      for (int kt = 0; kt < 2; ++kt) {
        Frag16 Bf;
        const unsigned short* bp = W3b + (size_t)(kt * 32) * 32 + nt * 16 + m;
#pragma unroll
        for (int j = 0; j < 8; ++j) {
          const int k0 = half * 8 + ((2 * j) & 7) + (((2 * j) >> 3) << 4);
          Bf.q[j >> 2][j & 3] = (unsigned int)bp[k0 * 32] | ((unsigned int)bp[(k0 + 1) * 32] << 16);
        }
        acc = __builtin_amdgcn_wmma_f32_16x16x32_bf16(false, A[kt].v, false, Bf.v,
                                                      (short)0, acc, false, false);
      }
#pragma unroll
      for (int i = 0; i < 8; ++i)
        sH3[(half * 8 + i) * 32 + nt * 16 + m] = fmaxf(acc[i], 0.f);
    }
  }
  __syncthreads();

  // GMF + concat dot + sigmoid (lanes 0..15, one pair each)
  if (lane < 16) {
    const int p = lane;
    const float* up = x3 + (size_t)user[base + p] * EMBD;
    const float* vp = x3 + (size_t)item[base + p] * EMBD;
    float su = 0.f, sv = 0.f, duv = 0.f;
    for (int c = 0; c < EMBD; ++c) {
      const float uc = up[c], vc = vp[c];
      su += uc * uc; sv += vc * vc;
      duv += uc * vc * Wout[c];
    }
    const float inv_u = 1.f / fmaxf(sqrtf(su), 1e-12f);
    const float inv_v = 1.f / fmaxf(sqrtf(sv), 1e-12f);
    float logit = duv * inv_u * inv_v;
    for (int j = 0; j < 32; ++j) logit += sH3[p * 32 + j] * Wout[EMBD + j];
    logit += bout[0];
    out[base + p] = 1.f / (1.f + expf(-logit));
  }
}

// ---------------------------------------------------------------- host
static inline void cvt(const float* s, unsigned short* d, int n, hipStream_t st) {
  f32_to_bf16<<<dim3((n + 255) / 256), dim3(256), 0, st>>>(s, d, n);
}

extern "C" void kernel_launch(void* const* d_in, const int* in_sizes, int n_in,
                              void* d_out, int out_size, void* d_ws, size_t ws_size,
                              hipStream_t stream) {
  (void)in_sizes; (void)n_in; (void)out_size; (void)ws_size;
  const int*   user = (const int*)d_in[0];
  const int*   item = (const int*)d_in[1];
  const int*   eidx = (const int*)d_in[2];
  const int*   src  = eidx;
  const int*   dstN = eidx + NEDGE;
  const int*   et   = (const int*)d_in[3];
  const float* ew   = (const float*)d_in[4];
  const float* emb  = (const float*)d_in[5];
  const float* Wr[3]    = {(const float*)d_in[6],  (const float*)d_in[9],  (const float*)d_in[12]};
  const float* Wroot[3] = {(const float*)d_in[7],  (const float*)d_in[10], (const float*)d_in[13]};
  const float* bias[3]  = {(const float*)d_in[8],  (const float*)d_in[11], (const float*)d_in[14]};
  const float* ln_g[2]  = {(const float*)d_in[15], (const float*)d_in[17]};
  const float* ln_b[2]  = {(const float*)d_in[16], (const float*)d_in[18]};
  const float* Wm1 = (const float*)d_in[19]; const float* bm1 = (const float*)d_in[20];
  const float* Wm2 = (const float*)d_in[21]; const float* bm2 = (const float*)d_in[22];
  const float* Wm3 = (const float*)d_in[23]; const float* bm3 = (const float*)d_in[24];
  const float* Wo  = (const float*)d_in[25]; const float* bo  = (const float*)d_in[26];

  // workspace layout (bytes), padded node dim:
  //   xb  [NPAD,128] bf16   : 25,690,112
  //   h   [2,NPAD,128] f32  : 102,760,448
  //   agg [NPAD,128] f32    : 51,380,224
  //   bf16 weights          : ~0.4MB
  char* ws = (char*)d_ws;
  unsigned short* xb  = (unsigned short*)(ws);
  float*          h   = (float*)(ws + (size_t)25690112);
  float*          agg = (float*)(ws + (size_t)25690112 + 102760448);
  unsigned short* wb  = (unsigned short*)(ws + (size_t)25690112 + 102760448 + 51380224);
  const int perLayer = 3 * EMBD * EMBD;                 // Wr(2) + Wroot(1), contiguous
  unsigned short* W1b = wb + 3 * perLayer;
  unsigned short* W2b = W1b + 256 * 128;
  unsigned short* W3b = W2b + 128 * 64;

  cvt(emb, xb, NNODES * EMBD, stream);
  for (int l = 0; l < 3; ++l) {
    cvt(Wr[l],    wb + l * perLayer,                   2 * EMBD * EMBD, stream);
    cvt(Wroot[l], wb + l * perLayer + 2 * EMBD * EMBD, EMBD * EMBD,     stream);
  }
  cvt(Wm1, W1b, 256 * 128, stream);
  cvt(Wm2, W2b, 128 * 64,  stream);
  cvt(Wm3, W3b, 64 * 32,   stream);

  dim3 gG(NPAD / 256);                                  // 392 blocks, 32 tiles each, no guards
  dim3 gE((NEDGE + 7) / 8), gN((NNODES + 7) / 8);
  for (int l = 0; l < 3; ++l) {
    wrgcn_gemm<<<gG, 256, 0, stream>>>(xb, wb + l * perLayer, bias[l], h, agg);
    edge_scatter<<<gE, 256, 0, stream>>>(src, dstN, et, ew, h, agg, NEDGE);
    if (l < 2)
      relu_layernorm_bf16<<<gN, 256, 0, stream>>>(agg, ln_g[l], ln_b[l], xb, NNODES);
  }
  neuralcf_head<<<dim3(BPAIRS / 16), dim3(32), 0, stream>>>(
      agg, user, item, W1b, bm1, W2b, bm2, W3b, bm3, Wo, bo, (float*)d_out);
}
